// RCNNHead_33715493274078
// MI455X (gfx1250) — matrix operand order repaired and checked by reading, not compiled, not run on hardware
//
#include <hip/hip_runtime.h>

// ---------------------------------------------------------------------------
// Problem constants (match reference setup_inputs)
// ---------------------------------------------------------------------------
#define BATCH   2
#define RPERB   512
#define NROI    (BATCH * RPERB)      // 1024
#define CCH     256
#define POOL    7
#define NPOS    (POOL * POOL)        // 49
#define DFEAT   (NPOS * CCH)         // 12544
#define H1      1024
#define NCLS    81
#define NBOX    (NCLS * 4)           // 324

typedef __attribute__((ext_vector_type(16))) __bf16         v16bf;
typedef __attribute__((ext_vector_type(8)))  float          v8f;
typedef __attribute__((ext_vector_type(8)))  unsigned short ushort8;

union FragBF16 {
    v16bf   v;
    ushort8 h[2];
};

__device__ __forceinline__ unsigned short f2bf(float f) {
    unsigned int u = __float_as_uint(f);
    unsigned int r = u + 0x7FFFu + ((u >> 16) & 1u);   // round-to-nearest-even
    return (unsigned short)(r >> 16);
}

// Low 32 bits of a generic pointer into __shared__ = LDS byte offset
// (flat-address LDS aperture truncation, ISA 10.2).
__device__ __forceinline__ unsigned lds_off(const void* p) {
    return (unsigned)(unsigned long long)p;
}

// CDNA5 async global->LDS copy, 16 B per lane, tracked by ASYNCcnt.
__device__ __forceinline__ void async_copy_b128(const void* gptr, unsigned ldsOff) {
    asm volatile("global_load_async_to_lds_b128 %0, %1, off"
                 :: "v"(ldsOff), "v"(gptr) : "memory");
}
__device__ __forceinline__ void wait_async_le4() {
    asm volatile("s_wait_asynccnt 0x4" ::: "memory");
}
__device__ __forceinline__ void wait_async_0() {
    asm volatile("s_wait_asynccnt 0x0" ::: "memory");
}

// ---------------------------------------------------------------------------
// fp32 -> bf16 conversion (weights)
// ---------------------------------------------------------------------------
__global__ void convert_bf16_kernel(const float* __restrict__ src,
                                    unsigned short* __restrict__ dst, int n) {
    int i = blockIdx.x * blockDim.x + threadIdx.x;
    int stride = gridDim.x * blockDim.x;
    for (; i < n; i += stride) dst[i] = f2bf(src[i]);
}

// ---------------------------------------------------------------------------
// Pyramid ROI align.  grid.x = NROI*NPOS, block = 256 (one thread per channel)
// Writes pooled features as bf16 in x[n][c*49 + ph*7 + pw] layout.
// ---------------------------------------------------------------------------
__global__ void roi_align_kernel(const float* __restrict__ p2,
                                 const float* __restrict__ p3,
                                 const float* __restrict__ p4,
                                 const float* __restrict__ p5,
                                 const float* __restrict__ rois,
                                 unsigned short* __restrict__ xout) {
    const int n   = blockIdx.x / NPOS;
    const int pos = blockIdx.x % NPOS;
    const int ph  = pos / POOL;
    const int pw  = pos % POOL;
    const int c   = threadIdx.x;           // 0..255

    const float y1 = rois[n * 4 + 0];
    const float x1 = rois[n * 4 + 1];
    const float y2 = rois[n * 4 + 2];
    const float x2 = rois[n * 4 + 3];

    // level selection: clip(floor(4 + log2(sqrt(h*w*1024*1024)/224)), 2, 5)
    const float h  = y2 - y1;
    const float w  = x2 - x1;
    float sz = sqrtf(h * w * 1048576.0f);
    sz = fmaxf(sz, 1e-6f);
    float lev = floorf(4.0f + log2f(sz / 224.0f));
    lev = fminf(fmaxf(lev, 2.0f), 5.0f);
    const int l = (int)lev - 2;            // 0..3 -> p2..p5

    const float* fm = (l == 0) ? p2 : (l == 1) ? p3 : (l == 2) ? p4 : p5;
    const int H = 256 >> l;                // W == H

    const int b = n / RPERB;

    const float gy = (float)ph * (1.0f / (POOL - 1));
    const float gx = (float)pw * (1.0f / (POOL - 1));
    const float ys = (y1 + gy * h) * (float)(H - 1);
    const float xs = (x1 + gx * w) * (float)(H - 1);

    const float y0f = floorf(ys);
    const float x0f = floorf(xs);
    const float wy = ys - y0f;
    const float wx = xs - x0f;
    int y0 = min(max((int)y0f, 0), H - 1);
    int x0 = min(max((int)x0f, 0), H - 1);
    int y1i = min(y0 + 1, H - 1);
    int x1i = min(x0 + 1, H - 1);

    const float* base = fm + ((size_t)b * CCH + c) * (size_t)H * H;
    const float v00 = base[(size_t)y0  * H + x0];
    const float v01 = base[(size_t)y0  * H + x1i];
    const float v10 = base[(size_t)y1i * H + x0];
    const float v11 = base[(size_t)y1i * H + x1i];

    const float top = v00 + (v01 - v00) * wx;
    const float bot = v10 + (v11 - v10) * wx;
    const float out = top + (bot - top) * wy;

    xout[(size_t)n * DFEAT + c * NPOS + pos] = f2bf(out);
}

// ---------------------------------------------------------------------------
// Shared WMMA tile geometry: BM=BN=128, BK=32; 8 waves; each wave 32x64.
// ---------------------------------------------------------------------------
#define BK   32
#define LDT  40   // LDS row stride in halves (80 B: 16-B aligned, bank-skewed)

// Fragment loaders (CDNA5 wave32 packings):
//  A: lane<16 -> row=lane,   halves K0-7 then K16-23
//     lane>=16-> row=lane-16,halves K8-15 then K24-31
//  B: lane<16 -> col=lane,   halves K0-7 then K16-23; lane>=16 -> col=lane-16, K8-15 then K24-31
__device__ __forceinline__ void load_fragA(FragBF16& f, const unsigned short* ls,
                                           int r, int hi) {
    f.h[0] = *reinterpret_cast<const ushort8*>(&ls[r * LDT + hi * 8]);
    f.h[1] = *reinterpret_cast<const ushort8*>(&ls[r * LDT + hi * 8 + 16]);
}
__device__ __forceinline__ void load_fragB(FragBF16& f, const unsigned short* ls,
                                           int r, int hi) {
    f.h[0] = *reinterpret_cast<const ushort8*>(&ls[r * LDT + hi * 8]);
    f.h[1] = *reinterpret_cast<const ushort8*>(&ls[r * LDT + hi * 8 + 16]);
}

// ---------------------------------------------------------------------------
// Async double-buffered bf16 WMMA GEMM (full tiles only):
//   out[M x Nrows] = A[M x K] * W[Nrows x K]^T + bias
// Requires: M % 128 == 0, Nrows % 128 == 0, K % 32 == 0.
// Pipeline: issue tile k+1 via global_load_async_to_lds_b128 into buffer^1,
// s_wait_asynccnt 4 (tile k done in this wave), barrier (done in all waves),
// ds_load fragments + 8 WMMA, barrier (buffer free for reuse).
// ---------------------------------------------------------------------------
template <bool RELU, bool OUTBF16>
__global__ __launch_bounds__(256)
void gemm_wmma_async_kernel(const unsigned short* __restrict__ A,
                            const unsigned short* __restrict__ W,
                            const float* __restrict__ bias,
                            void* __restrict__ out,
                            int M, int Nrows, int K) {
    __shared__ __align__(16) unsigned short lsA[2][128 * LDT];
    __shared__ __align__(16) unsigned short lsB[2][128 * LDT];

    const int tid  = threadIdx.x;
    const int lane = tid & 31;
    const int wid  = tid >> 5;
    const int waveM = wid & 3;   // 4 waves down M
    const int waveN = wid >> 2;  // 2 waves across N

    const int m0 = blockIdx.y * 128;
    const int n0 = blockIdx.x * 128;

    const int lrow = lane & 15;
    const int hi   = lane >> 4;

    v8f acc[2][4];
    {
        v8f z = 0.0f;
        for (int mi = 0; mi < 2; ++mi)
            for (int ni = 0; ni < 4; ++ni) acc[mi][ni] = z;
    }

    // per-thread async copy slots: row = tid>>1, 16-half chunk = tid&1
    const int fillRow  = tid >> 1;
    const int fillHalf = tid & 1;
    const unsigned short* gA = A + (size_t)(m0 + fillRow) * K + fillHalf * 16;
    const unsigned short* gB = W + (size_t)(n0 + fillRow) * K + fillHalf * 16;
    const unsigned dA0 = lds_off(&lsA[0][fillRow * LDT + fillHalf * 16]);
    const unsigned dA1 = lds_off(&lsA[1][fillRow * LDT + fillHalf * 16]);
    const unsigned dB0 = lds_off(&lsB[0][fillRow * LDT + fillHalf * 16]);
    const unsigned dB1 = lds_off(&lsB[1][fillRow * LDT + fillHalf * 16]);

    const int steps = K / BK;

    // prologue: tile 0 into buffer 0 (4 async ops per thread)
    {
        async_copy_b128(gA,     dA0);
        async_copy_b128(gA + 8, dA0 + 16);
        async_copy_b128(gB,     dB0);
        async_copy_b128(gB + 8, dB0 + 16);
    }

    for (int kt = 0; kt < steps; ++kt) {
        const int cur = kt & 1;

        if (kt + 1 < steps) {   // issue tile kt+1 into the other buffer
            const int koff = (kt + 1) * BK;
            const unsigned dA = cur ? dA0 : dA1;
            const unsigned dB = cur ? dB0 : dB1;
            async_copy_b128(gA + koff,     dA);
            async_copy_b128(gA + koff + 8, dA + 16);
            async_copy_b128(gB + koff,     dB);
            async_copy_b128(gB + koff + 8, dB + 16);
            wait_async_le4();   // tile kt complete (this wave)
        } else {
            wait_async_0();
        }
        __syncthreads();        // tile kt complete (all waves)

        const unsigned short* sa = lsA[cur];
        const unsigned short* sb = lsB[cur];

        FragBF16 fa[2];
        for (int mi = 0; mi < 2; ++mi)
            load_fragA(fa[mi], sa, waveM * 32 + mi * 16 + lrow, hi);
        FragBF16 fb[4];
        for (int ni = 0; ni < 4; ++ni)
            load_fragB(fb[ni], sb, waveN * 64 + ni * 16 + lrow, hi);

        for (int mi = 0; mi < 2; ++mi)
            for (int ni = 0; ni < 4; ++ni)
                acc[mi][ni] = __builtin_amdgcn_wmma_f32_16x16x32_bf16(
                    false, fa[mi].v, false, fb[ni].v,
                    (short)0, acc[mi][ni], false, false);

        __syncthreads();        // all waves done reading buffer `cur`
    }

    // epilogue: C layout VGPR j -> M = j + hi*8 ; N = lane&15
    for (int mi = 0; mi < 2; ++mi) {
        for (int ni = 0; ni < 4; ++ni) {
            const int col = n0 + waveN * 64 + ni * 16 + lrow;
            const float bcol = bias[col];
            for (int j = 0; j < 8; ++j) {
                const int row = m0 + waveM * 32 + mi * 16 + hi * 8 + j;
                float v = acc[mi][ni][j] + bcol;
                if (RELU) v = fmaxf(v, 0.0f);
                if (OUTBF16)
                    ((unsigned short*)out)[(size_t)row * Nrows + col] = f2bf(v);
                else
                    ((float*)out)[(size_t)row * Nrows + col] = v;
            }
        }
    }
}

// ---------------------------------------------------------------------------
// Synchronous guarded GEMM for ragged Nrows (cls: 81, bbox: 324 heads).
// ---------------------------------------------------------------------------
template <bool RELU, bool OUTBF16>
__global__ __launch_bounds__(256)
void gemm_wmma_kernel(const unsigned short* __restrict__ A,
                      const unsigned short* __restrict__ W,
                      const float* __restrict__ bias,
                      void* __restrict__ out,
                      int M, int Nrows, int K) {
    __shared__ __align__(16) unsigned short lsA[128 * LDT];
    __shared__ __align__(16) unsigned short lsB[128 * LDT];

    const int tid  = threadIdx.x;
    const int lane = tid & 31;
    const int wid  = tid >> 5;
    const int waveM = wid & 3;
    const int waveN = wid >> 2;

    const int m0 = blockIdx.y * 128;
    const int n0 = blockIdx.x * 128;

    const int lrow = lane & 15;
    const int hi   = lane >> 4;

    v8f acc[2][4];
    {
        v8f z = 0.0f;
        for (int mi = 0; mi < 2; ++mi)
            for (int ni = 0; ni < 4; ++ni) acc[mi][ni] = z;
    }

    const int fillRow  = tid >> 1;
    const int fillHalf = tid & 1;
    const ushort8 zero8 = (ushort8)0;
    const bool bValid = (n0 + fillRow) < Nrows;

    for (int k0 = 0; k0 < K; k0 += BK) {
        {
            const ushort8* g = reinterpret_cast<const ushort8*>(
                A + (size_t)(m0 + fillRow) * K + k0 + fillHalf * 16);
            ushort8* s = reinterpret_cast<ushort8*>(
                &lsA[fillRow * LDT + fillHalf * 16]);
            s[0] = g[0];
            s[1] = g[1];
        }
        {
            ushort8* s = reinterpret_cast<ushort8*>(
                &lsB[fillRow * LDT + fillHalf * 16]);
            if (bValid) {
                const ushort8* g = reinterpret_cast<const ushort8*>(
                    W + (size_t)(n0 + fillRow) * K + k0 + fillHalf * 16);
                s[0] = g[0];
                s[1] = g[1];
            } else {
                s[0] = zero8;
                s[1] = zero8;
            }
        }
        __syncthreads();

        FragBF16 fa[2];
        for (int mi = 0; mi < 2; ++mi)
            load_fragA(fa[mi], lsA, waveM * 32 + mi * 16 + lrow, hi);
        FragBF16 fb[4];
        for (int ni = 0; ni < 4; ++ni)
            load_fragB(fb[ni], lsB, waveN * 64 + ni * 16 + lrow, hi);

        for (int mi = 0; mi < 2; ++mi)
            for (int ni = 0; ni < 4; ++ni)
                acc[mi][ni] = __builtin_amdgcn_wmma_f32_16x16x32_bf16(
                    false, fa[mi].v, false, fb[ni].v,
                    (short)0, acc[mi][ni], false, false);

        __syncthreads();
    }

    for (int mi = 0; mi < 2; ++mi) {
        for (int ni = 0; ni < 4; ++ni) {
            const int col = n0 + waveN * 64 + ni * 16 + lrow;
            if (col >= Nrows) continue;
            const float bcol = bias[col];
            for (int j = 0; j < 8; ++j) {
                const int row = m0 + waveM * 32 + mi * 16 + hi * 8 + j;
                float v = acc[mi][ni][j] + bcol;
                if (RELU) v = fmaxf(v, 0.0f);
                if (OUTBF16)
                    ((unsigned short*)out)[(size_t)row * Nrows + col] = f2bf(v);
                else
                    ((float*)out)[(size_t)row * Nrows + col] = v;
            }
        }
    }
}

// ---------------------------------------------------------------------------
// Row softmax over NCLS=81 logits; one block (128 threads) per ROI.
// ---------------------------------------------------------------------------
__global__ void softmax_rows_kernel(const float* __restrict__ logits,
                                    float* __restrict__ probs) {
    __shared__ float red[128];
    const int row = blockIdx.x;
    const int t   = threadIdx.x;

    float v = (t < NCLS) ? logits[(size_t)row * NCLS + t] : -3.402823466e38f;
    red[t] = v;
    __syncthreads();
    for (int s = 64; s > 0; s >>= 1) {
        if (t < s) red[t] = fmaxf(red[t], red[t + s]);
        __syncthreads();
    }
    const float mx = red[0];
    __syncthreads();

    float e = (t < NCLS) ? __expf(v - mx) : 0.0f;
    red[t] = e;
    __syncthreads();
    for (int s = 64; s > 0; s >>= 1) {
        if (t < s) red[t] += red[t + s];
        __syncthreads();
    }
    const float inv = 1.0f / red[0];
    if (t < NCLS) probs[(size_t)row * NCLS + t] = e * inv;
}

// ---------------------------------------------------------------------------
// Launch
// ---------------------------------------------------------------------------
extern "C" void kernel_launch(void* const* d_in, const int* in_sizes, int n_in,
                              void* d_out, int out_size, void* d_ws, size_t ws_size,
                              hipStream_t stream) {
    const float* p2     = (const float*)d_in[0];
    const float* p3     = (const float*)d_in[1];
    const float* p4     = (const float*)d_in[2];
    const float* p5     = (const float*)d_in[3];
    const float* rois   = (const float*)d_in[4];
    const float* fc1_w  = (const float*)d_in[7];
    const float* fc1_b  = (const float*)d_in[8];
    const float* fc2_w  = (const float*)d_in[9];
    const float* fc2_b  = (const float*)d_in[10];
    const float* cls_w  = (const float*)d_in[11];
    const float* cls_b  = (const float*)d_in[12];
    const float* bbox_w = (const float*)d_in[13];
    const float* bbox_b = (const float*)d_in[14];

    // workspace carve-up (bf16 buffers), 256-B aligned
    char* ws = (char*)d_ws;
    size_t off = 0;
    auto take = [&](size_t halves) {
        size_t o = off;
        off += (halves * sizeof(unsigned short) + 255) & ~(size_t)255;
        return (unsigned short*)(ws + o);
    };
    unsigned short* x_bf  = take((size_t)NROI * DFEAT);
    unsigned short* w1_bf = take((size_t)H1 * DFEAT);
    unsigned short* w2_bf = take((size_t)H1 * H1);
    unsigned short* wc_bf = take((size_t)NCLS * H1);
    unsigned short* wb_bf = take((size_t)NBOX * H1);
    unsigned short* h1_bf = take((size_t)NROI * H1);
    unsigned short* h2_bf = take((size_t)NROI * H1);
    (void)ws_size; (void)in_sizes; (void)n_in; (void)out_size;

    float* logits = (float*)d_out;
    float* probs  = logits + (size_t)NROI * NCLS;
    float* bbox   = probs  + (size_t)NROI * NCLS;

    // 1) weight conversion fp32 -> bf16
    {
        const int T = 256;
        convert_bf16_kernel<<<2048, T, 0, stream>>>(fc1_w,  w1_bf, H1 * DFEAT);
        convert_bf16_kernel<<<1024, T, 0, stream>>>(fc2_w,  w2_bf, H1 * H1);
        convert_bf16_kernel<<<128,  T, 0, stream>>>(cls_w,  wc_bf, NCLS * H1);
        convert_bf16_kernel<<<256,  T, 0, stream>>>(bbox_w, wb_bf, NBOX * H1);
    }

    // 2) pyramid ROI align -> x (bf16, [NROI x DFEAT])
    roi_align_kernel<<<NROI * NPOS, CCH, 0, stream>>>(p2, p3, p4, p5, rois, x_bf);

    // 3) fc1: [1024 x 12544] x [1024 x 12544]^T -> relu -> h1 (bf16)  [async]
    gemm_wmma_async_kernel<true, true><<<dim3(H1 / 128, NROI / 128), 256, 0, stream>>>(
        x_bf, w1_bf, fc1_b, h1_bf, NROI, H1, DFEAT);

    // 4) fc2: [1024 x 1024] x [1024 x 1024]^T -> relu -> h2 (bf16)  [async]
    gemm_wmma_async_kernel<true, true><<<dim3(H1 / 128, NROI / 128), 256, 0, stream>>>(
        h1_bf, w2_bf, fc2_b, h2_bf, NROI, H1, H1);

    // 5) cls head: [1024 x 1024] x [81 x 1024]^T -> logits (fp32, d_out)
    gemm_wmma_kernel<false, false><<<dim3(1, NROI / 128), 256, 0, stream>>>(
        h2_bf, wc_bf, cls_b, logits, NROI, NCLS, H1);

    // 6) bbox head: [1024 x 1024] x [324 x 1024]^T -> bbox (fp32, d_out)
    gemm_wmma_kernel<false, false><<<dim3(3, NROI / 128), 256, 0, stream>>>(
        h2_bf, wb_bf, bbox_b, bbox, NROI, NBOX, H1);

    // 7) softmax over the 81 logits per ROI
    softmax_rows_kernel<<<NROI, 128, 0, stream>>>(logits, probs);
}
